// InsaneMambaModel_62448824484282
// MI455X (gfx1250) — compile-verified, matrix operand order checked by reading, not run on hardware
//
#include <hip/hip_runtime.h>
#include <hip/hip_bf16.h>
#include <math.h>

// ---------------------------------------------------------------------------
// InsaneMambaModel on MI455X (gfx1250, wave32, WMMA).
// L==1 everywhere => each mamba block reduces to GEMM / elementwise / GEMM.
// All GEMMs use v_wmma_f32_16x16x32_bf16 (f32 accumulate).
// Round 4: weights pre-converted f32->bf16 ONCE into d_ws (reused by all
// 1024 workgroups) — kills the v_cvt_pk_bf16_f32 chains in the hot GEMM
// loops, halves B-side global bytes, and removes the n<nmax guard via
// zero-padded xproj (36 -> 48 rows).
// ---------------------------------------------------------------------------

typedef __attribute__((ext_vector_type(16))) __bf16 v16bf;
typedef __attribute__((ext_vector_type(8)))  float  v8f;

// K-pair base index held by fragment VGPR v (0..7) for lane-half h (0/1).
// Matches CDNA5 ISA 16-bit A 16x32 layout: V0:K0,1|K8,9  V4:K16,17|K24,25 ...
__device__ __forceinline__ int kpb(int v, int h) {
  return ((v & 4) << 2) + (h << 3) + ((v & 3) << 1);
}

// A fragment (16xK tile, row-major bf16 in LDS, leading dim ld, K offset k0).
__device__ __forceinline__ v16bf load_a_lds(const __bf16* A, int ld, int k0) {
  const int lane = threadIdx.x & 31;
  const int m = lane & 15, h = lane >> 4;
  const __bf16* row = A + m * ld + k0;
  v16bf r;
#pragma unroll
  for (int v = 0; v < 8; ++v) {
    const int k = kpb(v, h);
    r[2 * v]     = row[k];
    r[2 * v + 1] = row[k + 1];
  }
  return r;
}

// B fragment (Kx16 tile) from a pre-converted bf16 weight stored [N][K]
// row-major.  Per lane this is two 16B chunks => 2x global_load_b128.
__device__ __forceinline__ v16bf load_b_gbf16(const __bf16* __restrict__ W,
                                              int n0, int k0, int ldk) {
  const int lane = threadIdx.x & 31;
  const int n = n0 + (lane & 15), h = lane >> 4;
  const __bf16* row = W + (size_t)n * ldk + k0;
  v16bf r;
#pragma unroll
  for (int v = 0; v < 8; ++v) {
    const int k = kpb(v, h);
    r[2 * v]     = row[k];
    r[2 * v + 1] = row[k + 1];
  }
  return r;
}

#define WMMA_BF16(a, b, c) \
  __builtin_amdgcn_wmma_f32_16x16x32_bf16(false, (a), false, (b), (short)0, (c), false, false)

// Fast transcendentals: v_exp_f32 / v_rcp_f32 / v_log_f32, no IEEE div ladder.
__device__ __forceinline__ float sigmoid_fast(float x) {
  return __builtin_amdgcn_rcpf(1.0f + __expf(-x));
}
__device__ __forceinline__ float silu_fast(float x) { return x * sigmoid_fast(x); }
__device__ __forceinline__ float softplus_fast(float x) {
  return (x > 20.0f) ? x : __logf(1.0f + __expf(x));
}

// ---------------------------------------------------------------------------
// Weight pre-conversion kernels (run once per launch; weights are reused by
// ~1024 workgroups, so on-the-fly cvt in the GEMM loops is pure waste).
// ---------------------------------------------------------------------------
__global__ void __launch_bounds__(256) cvt_f32_bf16_kernel(
    const float* __restrict__ src, __bf16* __restrict__ dst, int n) {
  const int i = blockIdx.x * 256 + threadIdx.x;
  if (i < n) dst[i] = (__bf16)src[i];
}

// branch xproj_w [4][36][128] -> [4][48][128], rows 36..47 zero-padded.
__global__ void __launch_bounds__(256) cvt_pad_xproj_kernel(
    const float* __restrict__ src, __bf16* __restrict__ dst) {
  const int i = blockIdx.x * 256 + threadIdx.x;
  if (i < 4 * 48 * 128) {
    const int k = i & 127, r = (i >> 7) % 48, f = i / (48 * 128);
    dst[i] = (r < 36) ? (__bf16)src[(f * 36 + r) * 128 + k] : (__bf16)0.0f;
  }
}

// ---------------------------------------------------------------------------
// Kernel 1: 4 branch mamba blocks (d_model=64, di=128, n=16, dt_rank=4).
// 128 threads = 4 waves, wave f owns branch f for a 16-row batch tile.
// Writes combined[b, f*64+d] as bf16 into d_ws.
// ---------------------------------------------------------------------------
#define BR_REGION 15424  // xc 4096 + zs 4096 + xdb 3072 + proj/y 4096 + s 64

__global__ void __launch_bounds__(128) mamba_branch_kernel(
    const float* __restrict__ x, const int* __restrict__ fidx,
    const float* __restrict__ proj_w, const float* __restrict__ proj_b,
    const float* __restrict__ bD,     const float* __restrict__ b_convb,
    const float* __restrict__ b_convw,const float* __restrict__ b_dtb,
    const float* __restrict__ b_dtw,
    const __bf16* __restrict__ b_inw, const __bf16* __restrict__ b_outw,
    const __bf16* __restrict__ b_xproj,   // padded [4][48][128]
    __bf16* __restrict__ combined) {
  __shared__ char smem[4 * BR_REGION];  // 61,696 B
  const int lane = threadIdx.x & 31;
  const int f    = threadIdx.x >> 5;   // branch index
  const int b0   = blockIdx.x * 16;
  const int nn   = lane & 15, hh = lane >> 4;

  char* base = smem + f * BR_REGION;
  __bf16* xcL  = (__bf16*)(base);            // [16][128] silu(conv(xi))
  __bf16* zsL  = (__bf16*)(base + 4096);     // [16][128] silu(z)
  float*  xdbL = (float* )(base + 8192);     // [16][48]  (36 valid cols)
  __bf16* pyL  = (__bf16*)(base + 11264);    // proj [16][64], later y [16][128]
  float*  sL   = (float* )(base + 15360);    // [16] Bm.Cm per row

  __builtin_prefetch(b_inw + (size_t)f * 256 * 64 + lane * 128, 0, 3);

  // P0: proj[b,d] = x[b, fidx[f]] * proj_w[f,d] + proj_b[f,d]
  const int fi = fidx[f];
  for (int i = lane; i < 16 * 64; i += 32) {
    const int m = i >> 6, d = i & 63;
    const float fv = x[(size_t)(b0 + m) * 8 + fi];
    pyL[m * 64 + d] = (__bf16)(fv * proj_w[f * 64 + d] + proj_b[f * 64 + d]);
  }
  __syncthreads();

  // P1: xz = proj @ in_w^T   [16,64] x [64 -> 256], conv/silu epilogue.
  // half 0 => xi tiles (n<128), half 1 => z tiles; A fragment shared by the
  // 8 n-tiles of each half (multi-accumulator loop interchange).
  const __bf16* in_w = b_inw + (size_t)f * 256 * 64;
  for (int half = 0; half < 2; ++half) {
    v8f acc[8] = {};
    for (int kt = 0; kt < 2; ++kt) {
      v16bf a = load_a_lds(pyL, 64, kt * 32);
#pragma unroll
      for (int it = 0; it < 8; ++it) {
        v16bf b = load_b_gbf16(in_w, half * 128 + it * 16, kt * 32, 64);
        acc[it] = WMMA_BF16(a, b, acc[it]);
      }
    }
    if (half == 0) {
#pragma unroll
      for (int it = 0; it < 8; ++it) {
        const int n = it * 16 + nn;
        const float cw3 = b_convw[(size_t)f * 512 + n * 4 + 3];
        const float cb  = b_convb[f * 128 + n];
#pragma unroll
        for (int r = 0; r < 8; ++r)
          xcL[(r + 8 * hh) * 128 + n] = (__bf16)silu_fast(acc[it][r] * cw3 + cb);
      }
    } else {
#pragma unroll
      for (int it = 0; it < 8; ++it) {
        const int n = it * 16 + nn;
#pragma unroll
        for (int r = 0; r < 8; ++r)
          zsL[(r + 8 * hh) * 128 + n] = (__bf16)silu_fast(acc[it][r]);
      }
    }
  }
  __syncthreads();

  // P2: xdb = xc @ xproj_w^T   [16,128] x [128 -> 48] (rows 36+ pre-zeroed)
  const __bf16* xproj = b_xproj + (size_t)f * 48 * 128;
  {
    v8f acc[3] = {};
    for (int kt = 0; kt < 4; ++kt) {
      v16bf a = load_a_lds(xcL, 128, kt * 32);
#pragma unroll
      for (int it = 0; it < 3; ++it) {
        v16bf b = load_b_gbf16(xproj, it * 16, kt * 32, 128);
        acc[it] = WMMA_BF16(a, b, acc[it]);
      }
    }
#pragma unroll
    for (int it = 0; it < 3; ++it)
#pragma unroll
      for (int r = 0; r < 8; ++r)
        xdbL[(r + 8 * hh) * 48 + it * 16 + nn] = acc[it][r];
  }
  __syncthreads();

  // P3: s[m] = Bm . Cm   (cols 4..19 and 20..35)
  if (lane < 16) {
    float s = 0.0f;
#pragma unroll
    for (int n = 0; n < 16; ++n)
      s += xdbL[lane * 48 + 4 + n] * xdbL[lane * 48 + 20 + n];
    sL[lane] = s;
  }
  __syncthreads();

  // P4: y = (softplus(dt_w@xdb + dt_b) * xc * s + xc * D) * silu(z)
  for (int i = lane; i < 16 * 128; i += 32) {
    const int m = i >> 7, e = i & 127;
    float dtv = b_dtb[f * 128 + e];
#pragma unroll
    for (int r = 0; r < 4; ++r)
      dtv += b_dtw[(size_t)f * 512 + e * 4 + r] * xdbL[m * 48 + r];
    dtv = softplus_fast(dtv);
    const float xcv = (float)xcL[m * 128 + e];
    const float yv  = (dtv * xcv * sL[m] + xcv * bD[f * 128 + e]) *
                      (float)zsL[m * 128 + e];
    pyL[m * 128 + e] = (__bf16)yv;
  }
  __syncthreads();

  // P5: out = y @ out_w^T   [16,128] x [128 -> 64] -> combined[:, f*64:]
  const __bf16* out_w = b_outw + (size_t)f * 64 * 128;
  {
    v8f acc[4] = {};
    for (int kt = 0; kt < 4; ++kt) {
      v16bf a = load_a_lds(pyL, 128, kt * 32);
#pragma unroll
      for (int it = 0; it < 4; ++it) {
        v16bf b = load_b_gbf16(out_w, it * 16, kt * 32, 128);
        acc[it] = WMMA_BF16(a, b, acc[it]);
      }
    }
#pragma unroll
    for (int it = 0; it < 4; ++it)
#pragma unroll
      for (int r = 0; r < 8; ++r)
        combined[(size_t)(b0 + r + 8 * hh) * 256 + f * 64 + it * 16 + nn] =
            (__bf16)acc[it][r];
  }
}

// ---------------------------------------------------------------------------
// Kernel 2: global mamba (d_model=256, di=512, n=16, dt_rank=16) + MLP head.
// 256 threads = 8 waves per 16-row batch tile.
// ---------------------------------------------------------------------------
__global__ void __launch_bounds__(256) mamba_global_head_kernel(
    const __bf16* __restrict__ combined,
    const float* __restrict__ gD,     const float* __restrict__ g_convb,
    const float* __restrict__ g_convw,const float* __restrict__ g_dtb,
    const float* __restrict__ g_dtw,
    const __bf16* __restrict__ g_inw, const __bf16* __restrict__ g_outw,
    const __bf16* __restrict__ g_xproj, const __bf16* __restrict__ head_w1,
    const float* __restrict__ head_b1,
    const float* __restrict__ head_w2,const float* __restrict__ head_b2,
    float* __restrict__ out) {
  __shared__ char smem[52288];
  __bf16* AY   = (__bf16*)(smem);          // A_in [16][256] bf16, later y [16][512]
  __bf16* XG   = (__bf16*)(smem + 16384);  // xc [16][512],  later g [16][256]
  __bf16* ZSf  = (__bf16*)(smem + 32768);  // zs [16][512]
  float*  Hf   = (float* )(smem + 32768);  // later h [16][64] f32
  float*  xdbL = (float* )(smem + 49152);  // [16][48]
  float*  sL   = (float* )(smem + 52224);  // [16]

  const int tid  = threadIdx.x;
  const int lane = tid & 31, wave = tid >> 5;
  const int nn   = lane & 15, hh = lane >> 4;
  const int b0   = blockIdx.x * 16;

  __builtin_prefetch(g_inw + (size_t)tid * 1024, 0, 3);  // warm in_w via L2

  // P0: stage combined tile (already bf16 in ws)
  for (int i = tid; i < 16 * 256; i += 256)
    AY[i] = combined[(size_t)b0 * 256 + i];
  __syncthreads();

  // P1: xz = A @ in_w^T  [16,256] x [256 -> 1024]; 8 n-tiles per wave.
  // Wave w owns n-tiles (w + it*8): it<4 are xi (n<512), it>=4 are z.
  // One A fragment per k-step, shared by all 8 accumulators.
  {
    v8f acc[8] = {};
    for (int kt = 0; kt < 8; ++kt) {
      v16bf a = load_a_lds(AY, 256, kt * 32);
#pragma unroll
      for (int it = 0; it < 8; ++it) {
        const int n0 = (wave + it * 8) * 16;
        v16bf b = load_b_gbf16(g_inw, n0, kt * 32, 256);
        acc[it] = WMMA_BF16(a, b, acc[it]);
      }
    }
#pragma unroll
    for (int it = 0; it < 8; ++it) {
      const int n = (wave + it * 8) * 16 + nn;
      if (it < 4) {  // compile-time: xi half
        const float cw3 = g_convw[n * 4 + 3];
        const float cb  = g_convb[n];
#pragma unroll
        for (int r = 0; r < 8; ++r)
          XG[(r + 8 * hh) * 512 + n] = (__bf16)silu_fast(acc[it][r] * cw3 + cb);
      } else {       // z half
#pragma unroll
        for (int r = 0; r < 8; ++r)
          ZSf[(r + 8 * hh) * 512 + (n - 512)] = (__bf16)silu_fast(acc[it][r]);
      }
    }
  }
  __syncthreads();

  // P2: xdb = xc @ xproj_w^T  [16,512] x [512 -> 48]; waves 0..2
  if (wave < 3) {
    const int n0 = wave * 16;
    v8f acc = {};
    for (int kt = 0; kt < 16; ++kt) {
      v16bf a = load_a_lds(XG, 512, kt * 32);
      v16bf b = load_b_gbf16(g_xproj, n0, kt * 32, 512);
      acc = WMMA_BF16(a, b, acc);
    }
#pragma unroll
    for (int r = 0; r < 8; ++r) xdbL[(r + 8 * hh) * 48 + n0 + nn] = acc[r];
  }
  __syncthreads();

  // P3: s[m] = Bm . Cm (cols 16..31 and 32..47)
  if (tid < 16) {
    float s = 0.0f;
#pragma unroll
    for (int n = 0; n < 16; ++n)
      s += xdbL[tid * 48 + 16 + n] * xdbL[tid * 48 + 32 + n];
    sL[tid] = s;
  }
  __syncthreads();

  // P4: y (overwrites AY region)
  for (int i = tid; i < 16 * 512; i += 256) {
    const int m = i >> 9, e = i & 511;
    float dtv = g_dtb[e];
#pragma unroll
    for (int r = 0; r < 16; ++r) dtv += g_dtw[e * 16 + r] * xdbL[m * 48 + r];
    dtv = softplus_fast(dtv);
    const float xcv = (float)XG[m * 512 + e];
    const float yv  = (dtv * xcv * sL[m] + xcv * gD[e]) * (float)ZSf[m * 512 + e];
    AY[m * 512 + e] = (__bf16)yv;
  }
  __syncthreads();

  // P5: g = y @ out_w^T  [16,512] x [512 -> 256] -> XG as bf16.
  // 2 n-tiles per wave, A fragment shared.
  {
    v8f acc0 = {}, acc1 = {};
    const int na = wave * 16, nb = (wave + 8) * 16;
    for (int kt = 0; kt < 16; ++kt) {
      v16bf a  = load_a_lds(AY, 512, kt * 32);
      v16bf b0 = load_b_gbf16(g_outw, na, kt * 32, 512);
      acc0 = WMMA_BF16(a, b0, acc0);
      v16bf b1 = load_b_gbf16(g_outw, nb, kt * 32, 512);
      acc1 = WMMA_BF16(a, b1, acc1);
    }
#pragma unroll
    for (int r = 0; r < 8; ++r) {
      XG[(r + 8 * hh) * 256 + na + nn] = (__bf16)acc0[r];
      XG[(r + 8 * hh) * 256 + nb + nn] = (__bf16)acc1[r];
    }
  }
  __syncthreads();

  // P6: h = relu(g @ head_w1^T + b1)  [16,256] x [256 -> 64]; waves 0..3
  if (wave < 4) {
    const int n0 = wave * 16;
    v8f acc = {};
    for (int kt = 0; kt < 8; ++kt) {
      v16bf a = load_a_lds(XG, 256, kt * 32);
      v16bf b = load_b_gbf16(head_w1, n0, kt * 32, 256);
      acc = WMMA_BF16(a, b, acc);
    }
#pragma unroll
    for (int r = 0; r < 8; ++r)
      Hf[(r + 8 * hh) * 64 + n0 + nn] = fmaxf(acc[r] + head_b1[n0 + nn], 0.0f);
  }
  __syncthreads();

  // P7: out = sigmoid(h . head_w2 + b2)
  if (tid < 16) {
    float acc = head_b2[0];
#pragma unroll 8
    for (int j = 0; j < 64; ++j) acc += Hf[tid * 64 + j] * head_w2[j];
    out[b0 + tid] = sigmoid_fast(acc);
  }
}

// ---------------------------------------------------------------------------
// d_in order: setup_inputs() insertion order; nested param dicts flatten in
// jax pytree (sorted-key) order: A_log, D, conv_b, conv_w, dt_b, dt_w, in_w,
// out_w, xproj_w.  (A_log is dead code at L==1.)
// ---------------------------------------------------------------------------
extern "C" void kernel_launch(void* const* d_in, const int* in_sizes, int n_in,
                              void* d_out, int out_size, void* d_ws, size_t ws_size,
                              hipStream_t stream) {
  (void)n_in; (void)out_size; (void)ws_size;
  const float* x       = (const float*)d_in[0];
  const int*   fidx    = (const int*  )d_in[1];
  const float* proj_w  = (const float*)d_in[2];
  const float* proj_b  = (const float*)d_in[3];
  const float* bD      = (const float*)d_in[5];
  const float* b_convb = (const float*)d_in[6];
  const float* b_convw = (const float*)d_in[7];
  const float* b_dtb   = (const float*)d_in[8];
  const float* b_dtw   = (const float*)d_in[9];
  const float* b_inw   = (const float*)d_in[10];
  const float* b_outw  = (const float*)d_in[11];
  const float* b_xproj = (const float*)d_in[12];
  const float* gD      = (const float*)d_in[14];
  const float* g_convb = (const float*)d_in[15];
  const float* g_convw = (const float*)d_in[16];
  const float* g_dtb   = (const float*)d_in[17];
  const float* g_dtw   = (const float*)d_in[18];
  const float* g_inw   = (const float*)d_in[19];
  const float* g_outw  = (const float*)d_in[20];
  const float* g_xproj = (const float*)d_in[21];
  const float* head_w1 = (const float*)d_in[22];
  const float* head_b1 = (const float*)d_in[23];
  const float* head_w2 = (const float*)d_in[24];
  const float* head_b2 = (const float*)d_in[25];

  const int Bsz = in_sizes[0] / 8;          // N_FEAT = 8

  // ws layout (bf16 elements): combined, then pre-converted weights.
  __bf16* wsb = (__bf16*)d_ws;
  size_t off = 0;
  __bf16* combined   = wsb + off; off += (size_t)Bsz * 256;   // 8 MB
  __bf16* b_inw_c    = wsb + off; off += 4 * 256 * 64;        // 65536
  __bf16* b_xproj_c  = wsb + off; off += 4 * 48 * 128;        // 24576 (padded)
  __bf16* b_outw_c   = wsb + off; off += 4 * 64 * 128;        // 32768
  __bf16* g_inw_c    = wsb + off; off += 1024 * 256;          // 262144
  __bf16* g_xproj_c  = wsb + off; off += 48 * 512;            // 24576
  __bf16* g_outw_c   = wsb + off; off += 256 * 512;           // 131072
  __bf16* head_w1_c  = wsb + off; off += 64 * 256;            // 16384

  // One-time weight conversions (graph-capturable, deterministic).
  #define CVT(src, dst, n) \
    cvt_f32_bf16_kernel<<<dim3(((n) + 255) / 256), dim3(256), 0, stream>>>((src), (dst), (n))
  CVT(b_inw,   b_inw_c,   4 * 256 * 64);
  CVT(b_outw,  b_outw_c,  4 * 64 * 128);
  CVT(g_inw,   g_inw_c,   1024 * 256);
  CVT(g_xproj, g_xproj_c, 48 * 512);
  CVT(g_outw,  g_outw_c,  256 * 512);
  CVT(head_w1, head_w1_c, 64 * 256);
  #undef CVT
  cvt_pad_xproj_kernel<<<dim3((4 * 48 * 128 + 255) / 256), dim3(256), 0, stream>>>(
      b_xproj, b_xproj_c);

  mamba_branch_kernel<<<dim3(Bsz / 16), dim3(128), 0, stream>>>(
      x, fidx, proj_w, proj_b, bD, b_convb, b_convw, b_dtb, b_dtw,
      b_inw_c, b_outw_c, b_xproj_c, combined);

  mamba_global_head_kernel<<<dim3(Bsz / 16), dim3(256), 0, stream>>>(
      combined, gD, g_convb, g_convw, g_dtb, g_dtw, g_inw_c, g_outw_c,
      g_xproj_c, head_w1_c, head_b1, head_w2, head_b2, (float*)d_out);
}
